// MultiTopologySurrogate_32238024524511
// MI455X (gfx1250) — matrix operand order repaired and checked by reading, not compiled, not run on hardware
//
#include <hip/hip_runtime.h>
#include <hip/hip_bf16.h>
#include <math.h>

#define NB 8192   // batch
#define HID 512

typedef __attribute__((ext_vector_type(16))) _Float16 v16h;
typedef __attribute__((ext_vector_type(8)))  _Float16 v8h;
typedef __attribute__((ext_vector_type(8)))  float    v8f;
typedef __attribute__((ext_vector_type(4)))  unsigned int v4u;
typedef __attribute__((ext_vector_type(8)))  int v8i;
typedef __attribute__((ext_vector_type(4)))  int v4i;

__constant__ float c_low[6]  = {1e-05f, 4.7e-05f, 2.0f, 5.0f, 50000.0f, 0.2f};
__constant__ float c_high[6] = {0.00047f, 0.001f, 100.0f, 24.0f, 500000.0f, 0.8f};
__constant__ int   c_islog[6] = {1, 1, 0, 0, 1, 0};

__device__ __forceinline__ float gelu_f(float x) {
  return 0.5f * x * (1.0f + erff(x * 0.70710678118654752f));
}

// A fragment: 16x32 f16, row-major source. Lane<16 holds row (lane&15),
// K = 0..15; lane>=16 holds K = 16..31 (contiguous, matches B).
__device__ __forceinline__ v16h load_a_frag(const _Float16* arow, int k0) {
  v8h lo = *(const v8h*)(arow + k0);
  v8h hi = *(const v8h*)(arow + k0 + 8);
  v16h a;
#pragma unroll
  for (int e = 0; e < 8; ++e) { a[e] = lo[e]; a[8 + e] = hi[e]; }
  return a;
}

// ---------------------------------------------------------------------------
// TDM: stage a [64][K] f16 tile (row stride K elements) from global into LDS.
// D# packing per CDNA5 ISA 8.3/8.4: group0 = {count|flags, lds_addr,
// global_addr[31:0], global_addr[56:32] | type=2<<30}; group1 holds data_size,
// tensor_dim0/1, tile_dim0/1/2, tensor_dim0_stride. 2-D tensor -> groups 2/3 = 0.
// This toolchain exposes the 6-arg builtin form (extra int32x8, zero-filled).
// Issued by wave 0 only; completion via TENSORcnt then workgroup barrier.
// ---------------------------------------------------------------------------
__device__ __forceinline__ void tdm_stage_tile_64xK(const _Float16* gsrc, _Float16* lds_dst, int K) {
  unsigned ldsa = (unsigned)(size_t)lds_dst;               // low 32 bits = LDS byte addr
  unsigned long long ga = (unsigned long long)(size_t)gsrc; // global byte addr
  v4u g0;
  g0[0] = 1u;                                              // count=1, no gather/restore
  g0[1] = ldsa;
  g0[2] = (unsigned)(ga & 0xFFFFFFFFu);
  g0[3] = (unsigned)((ga >> 32) & 0x1FFFFFFu) | (2u << 30); // addr[56:32] | type=2
  unsigned uk = (unsigned)K;
  v8i g1;
  g1[0] = (int)(1u << 16);                                 // data_size = 1 -> 2 bytes
  g1[1] = (int)((uk & 0xFFFFu) << 16);                     // tensor_dim0[15:0] @ bits 63:48
  g1[2] = (int)((uk >> 16) | (64u << 16));                 // tensor_dim0[31:16], tensor_dim1[15:0]=64
  g1[3] = (int)((uk & 0xFFFFu) << 16);                     // tensor_dim1[31:16]=0, tile_dim0=K
  g1[4] = 64;                                              // tile_dim1=64, tile_dim2=0 (2-D)
  g1[5] = (int)uk;                                         // tensor_dim0_stride[31:0] = K
  g1[6] = 0;                                               // stride0[47:32]=0, stride1 lo=0
  g1[7] = 0;
  v4i g2 = {0, 0, 0, 0};
  v4i g3 = {0, 0, 0, 0};
  v8i g4 = {0, 0, 0, 0, 0, 0, 0, 0};                       // extra group (clang-23 form)
  __builtin_amdgcn_tensor_load_to_lds(g0, g1, g2, g3, g4, 0);
  __builtin_amdgcn_s_wait_tensorcnt((short)0);
}

// ---------------------------------------------------------------------------
// Weight convert+transpose: W [K][N] f32 row-major -> Wt [N][Kpad] f16 (pad 0)
// ---------------------------------------------------------------------------
__global__ void w_to_f16t(const float* __restrict__ W, _Float16* __restrict__ Wt,
                          int K, int N, int Kpad) {
  int idx = blockIdx.x * blockDim.x + threadIdx.x;
  int total = N * Kpad;
  if (idx >= total) return;
  int n = idx / Kpad;
  int k = idx - n * Kpad;
  Wt[idx] = (k < K) ? (_Float16)W[(size_t)k * N + n] : (_Float16)0.0f;
}

// ---------------------------------------------------------------------------
// Input prep: normalize params (log/linear), gather embedding, pad to 64 cols
// ---------------------------------------------------------------------------
__global__ void prep_kernel(const float* __restrict__ params, const float* __restrict__ emb,
                            const int* __restrict__ ids, _Float16* __restrict__ x16) {
  int b = blockIdx.x * blockDim.x + threadIdx.x;
  if (b >= NB) return;
  _Float16* xr = x16 + (size_t)b * 64;
#pragma unroll
  for (int j = 0; j < 6; ++j) {
    float pv = params[b * 6 + j];
    float t;
    if (c_islog[j])
      t = (logf(pv) - logf(c_low[j])) / (logf(c_high[j]) - logf(c_low[j]));
    else
      t = (pv - c_low[j]) / (c_high[j] - c_low[j]);
    t = fminf(fmaxf(t, 0.0f), 1.0f);
    xr[j] = (_Float16)t;
  }
  int id = ids[b];
#pragma unroll
  for (int j = 0; j < 32; ++j) xr[6 + j] = (_Float16)emb[id * 32 + j];
#pragma unroll
  for (int j = 38; j < 64; ++j) xr[j] = (_Float16)0.0f;
}

// ---------------------------------------------------------------------------
// f16 WMMA GEMM: C(MxN,f32) = A(MxK,f16,row) @ Bt(NxK,f16)^T + bias
// 256 thr = 8 waves; block tile 128x64; wave tile 32x32 (4 accumulators).
// B tile (64 x K) staged in LDS once per block by the TDM; A streamed global.
// ---------------------------------------------------------------------------
__global__ __launch_bounds__(256) void gemm_f16_wmma(
    const _Float16* __restrict__ A, const _Float16* __restrict__ Bt,
    const float* __restrict__ bias, float* __restrict__ C,
    int M, int N, int K, int lda) {
  extern __shared__ __align__(16) _Float16 s_b[];   // [64][K]
  int tid  = threadIdx.x;
  int wave = tid >> 5;
  int lane = tid & 31;
  int blockN = blockIdx.x * 64;
  int blockM = blockIdx.y * 128;

  if (wave == 0)
    tdm_stage_tile_64xK(Bt + (size_t)blockN * K, s_b, K);
  __syncthreads();

  int m0  = blockM + (wave >> 1) * 32;   // rows m0..m0+31 (two A frags)
  int n0l = (wave & 1) * 32;             // LDS-local cols (two B frags)

  int r0 = m0 + (lane & 15);
  const _Float16* arow0 = A + (size_t)r0 * lda + ((lane & 16) ? 16 : 0);
  const _Float16* arow1 = arow0 + (size_t)16 * lda;
  int nl0 = n0l + (lane & 15);
  int nl1 = nl0 + 16;
  const _Float16* b0p = s_b + (size_t)nl0 * K + ((lane & 16) ? 16 : 0);
  const _Float16* b1p = s_b + (size_t)nl1 * K + ((lane & 16) ? 16 : 0);

  v8f acc00{}, acc01{}, acc10{}, acc11{};
  for (int k0 = 0; k0 < K; k0 += 32) {
    v16h a0 = load_a_frag(arow0, k0);
    v16h a1 = load_a_frag(arow1, k0);
    v16h b0 = *(const v16h*)(b0p + k0);   // ds_load_b128 x2, contiguous K
    v16h b1 = *(const v16h*)(b1p + k0);
    acc00 = __builtin_amdgcn_wmma_f32_16x16x32_f16(false, a0, false, b0, (short)0, acc00, false, false);
    acc01 = __builtin_amdgcn_wmma_f32_16x16x32_f16(false, a0, false, b1, (short)0, acc01, false, false);
    acc10 = __builtin_amdgcn_wmma_f32_16x16x32_f16(false, a1, false, b0, (short)0, acc10, false, false);
    acc11 = __builtin_amdgcn_wmma_f32_16x16x32_f16(false, a1, false, b1, (short)0, acc11, false, false);
  }
  int nc0 = blockN + nl0;
  int nc1 = blockN + nl1;
  float bv0 = bias ? bias[nc0] : 0.0f;
  float bv1 = bias ? bias[nc1] : 0.0f;
  int rb0 = m0 + ((lane & 16) ? 8 : 0);  // C layout: VGPR v -> row rb+v
  int rb1 = rb0 + 16;
#pragma unroll
  for (int v = 0; v < 8; ++v) {
    C[(size_t)(rb0 + v) * N + nc0] = acc00[v] + bv0;
    C[(size_t)(rb0 + v) * N + nc1] = acc01[v] + bv1;
    C[(size_t)(rb1 + v) * N + nc0] = acc10[v] + bv0;
    C[(size_t)(rb1 + v) * N + nc1] = acc11[v] + bv1;
  }
}

// ---------------------------------------------------------------------------
// Topology head GEMM: grid.z = e in [0,3). Writes only rows with ids[row]==e.
// feat(B x 1024) @ WhT[e](512 x 1024)^T + bh[e] -> hsel16 (f16)
// ---------------------------------------------------------------------------
__global__ __launch_bounds__(256) void gemm_head_wmma(
    const _Float16* __restrict__ feat, const _Float16* __restrict__ WhT,
    const float* __restrict__ bh, const int* __restrict__ ids,
    _Float16* __restrict__ out16) {
  const int K = 1024, N = 512;
  extern __shared__ __align__(16) _Float16 s_b[];   // [64][1024] = 128 KB
  int e = blockIdx.z;
  const _Float16* Bt = WhT + (size_t)e * N * K;
  const float* bias  = bh + e * N;

  int tid  = threadIdx.x;
  int wave = tid >> 5;
  int lane = tid & 31;
  int blockN = blockIdx.x * 64;
  int blockM = blockIdx.y * 128;

  if (wave == 0)
    tdm_stage_tile_64xK(Bt + (size_t)blockN * K, s_b, K);
  __syncthreads();

  int m0  = blockM + (wave >> 1) * 32;
  int n0l = (wave & 1) * 32;

  int r0 = m0 + (lane & 15);
  const _Float16* arow0 = feat + (size_t)r0 * K + ((lane & 16) ? 16 : 0);
  const _Float16* arow1 = arow0 + (size_t)16 * K;
  int nl0 = n0l + (lane & 15);
  int nl1 = nl0 + 16;
  const _Float16* b0p = s_b + (size_t)nl0 * K + ((lane & 16) ? 16 : 0);
  const _Float16* b1p = s_b + (size_t)nl1 * K + ((lane & 16) ? 16 : 0);

  v8f acc00{}, acc01{}, acc10{}, acc11{};
  for (int k0 = 0; k0 < K; k0 += 32) {
    v16h a0 = load_a_frag(arow0, k0);
    v16h a1 = load_a_frag(arow1, k0);
    v16h b0 = *(const v16h*)(b0p + k0);
    v16h b1 = *(const v16h*)(b1p + k0);
    acc00 = __builtin_amdgcn_wmma_f32_16x16x32_f16(false, a0, false, b0, (short)0, acc00, false, false);
    acc01 = __builtin_amdgcn_wmma_f32_16x16x32_f16(false, a0, false, b1, (short)0, acc01, false, false);
    acc10 = __builtin_amdgcn_wmma_f32_16x16x32_f16(false, a1, false, b0, (short)0, acc10, false, false);
    acc11 = __builtin_amdgcn_wmma_f32_16x16x32_f16(false, a1, false, b1, (short)0, acc11, false, false);
  }
  int nc0 = blockN + nl0;
  int nc1 = blockN + nl1;
  float bv0 = bias[nc0];
  float bv1 = bias[nc1];
  int rb0 = m0 + ((lane & 16) ? 8 : 0);
  int rb1 = rb0 + 16;
#pragma unroll
  for (int v = 0; v < 8; ++v) {
    int row0 = rb0 + v, row1 = rb1 + v;
    if (ids[row0] == e) {
      out16[(size_t)row0 * N + nc0] = (_Float16)(acc00[v] + bv0);
      out16[(size_t)row0 * N + nc1] = (_Float16)(acc01[v] + bv1);
    }
    if (ids[row1] == e) {
      out16[(size_t)row1 * N + nc0] = (_Float16)(acc10[v] + bv0);
      out16[(size_t)row1 * N + nc1] = (_Float16)(acc11[v] + bv1);
    }
  }
}

// ---------------------------------------------------------------------------
// LayerNorm + exact GELU, f32 in -> f16 out. One block (256 thr) per row.
// ---------------------------------------------------------------------------
__global__ __launch_bounds__(256) void ln_gelu_f16(
    const float* __restrict__ in, const float* __restrict__ g,
    const float* __restrict__ be, _Float16* __restrict__ out, int W) {
  __shared__ float r1[256];
  __shared__ float r2[256];
  int row = blockIdx.x;
  int tid = threadIdx.x;
  const float* x = in + (size_t)row * W;
  float s = 0.0f, s2 = 0.0f;
  for (int i = tid; i < W; i += 256) { float v = x[i]; s += v; s2 += v * v; }
  r1[tid] = s; r2[tid] = s2;
  __syncthreads();
  for (int off = 128; off > 0; off >>= 1) {
    if (tid < off) { r1[tid] += r1[tid + off]; r2[tid] += r2[tid + off]; }
    __syncthreads();
  }
  float mean = r1[0] / (float)W;
  float var  = r2[0] / (float)W - mean * mean;
  float rstd = rsqrtf(var + 1e-5f);
  _Float16* o = out + (size_t)row * W;
  for (int i = tid; i < W; i += 256) {
    float v = (x[i] - mean) * rstd * g[i] + be[i];
    o[i] = (_Float16)gelu_f(v);
  }
}

__global__ void gelu_to_f16(const float* __restrict__ in, _Float16* __restrict__ out, int n) {
  int i = blockIdx.x * blockDim.x + threadIdx.x;
  if (i < n) out[i] = (_Float16)gelu_f(in[i]);
}

// ---------------------------------------------------------------------------
// Fused conv stack, one sample per block (128 thr = 4 waves), all in LDS.
//   conv1 1->32 k7 pad3 + gelu (VALU)
//   conv2 32->32 k5 pad2 + gelu : 64 WMMA tiles, K=160 as 5 taps x 32ch
//   conv3 32->1 k3 pad1, residual add, per-topology scale (VALU)
// LDS: s_c2 514*32 f32 | s_c1 516*32 f16 [pos][ch] | s_w2 5*32*32 f16 | s_x 520 f16
// ---------------------------------------------------------------------------
__global__ __launch_bounds__(128) void conv_fused(
    const float* __restrict__ wvec,
    const float* __restrict__ cw1, const float* __restrict__ cb1,
    const float* __restrict__ cw2, const float* __restrict__ cb2,
    const float* __restrict__ cw3, const float* __restrict__ cb3,
    const float* __restrict__ scales, const int* __restrict__ ids,
    float* __restrict__ out) {
  extern __shared__ __align__(16) char smem[];
  float*    s_c2 = (float*)smem;                              // 514*32*4 = 65792 B
  _Float16* s_c1 = (_Float16*)(smem + 65792);                 // 516*32*2 = 33024 B
  _Float16* s_w2 = (_Float16*)(smem + 65792 + 33024);         // 5120*2   = 10240 B
  _Float16* s_x  = (_Float16*)(smem + 65792 + 33024 + 10240); // 520*2    = 1040  B

  int b   = blockIdx.x;
  int tid = threadIdx.x;
  const float* wr = wvec + (size_t)b * HID;

  // stage 0: input row -> f16 with halo (x[p] at s_x[p+3]; p in [-3,516))
  for (int p = tid; p < HID; p += 128) s_x[p + 3] = (_Float16)wr[p];
  if (tid < 8) { int z = (tid < 3) ? tid : (512 + tid); s_x[z] = (_Float16)0.0f; }
  // conv2 weights -> LDS f16 [t][o][c] (B-fragment needs contiguous c per o)
  for (int i = tid; i < 5 * 32 * 32; i += 128) {
    int t = i >> 10, o = (i >> 5) & 31, c = i & 31;
    s_w2[i] = (_Float16)cw2[(o * 32 + c) * 5 + t];
  }
  __syncthreads();

  // stage 1: conv1 + gelu -> s_c1[(p+2)*32 + o], p in [-2,514)
  for (int i = tid; i < 32 * HID; i += 128) {
    int o = i >> 9, p = i & 511;
    float acc = cb1[o];
#pragma unroll
    for (int t = 0; t < 7; ++t) acc += cw1[o * 7 + t] * (float)s_x[p + t];
    s_c1[(p + 2) * 32 + o] = (_Float16)gelu_f(acc);
  }
  { // zero halo rows p = -2,-1,512,513  (exactly 128 entries)
    static const int zr[4] = {0, 1, 514, 515};
    s_c1[zr[tid >> 5] * 32 + (tid & 31)] = (_Float16)0.0f;
  }
  __syncthreads();

  // stage 2: conv2 via WMMA. 64 tiles (32 pos-tiles x 2 outch-tiles), 4 waves.
  int wave = tid >> 5, lane = tid & 31;
  int prow = (lane & 15);
  int cb   = (lane & 16) ? 16 : 0;   // A: channel sub-range per lane half
  int kb   = (lane & 16) ? 16 : 0;   // B: K (channel) range per lane half
  for (int tile = wave; tile < 64; tile += 4) {
    int p0 = (tile >> 1) * 16;
    int n0 = (tile & 1) * 16;
    int oc = n0 + (lane & 15);
    v8f acc{};
#pragma unroll
    for (int t = 0; t < 5; ++t) {
      const _Float16* ap = s_c1 + (p0 + prow + t) * 32 + cb;   // pos p+t-2 shifted by halo
      v8h lo = *(const v8h*)ap;            // channels cb..cb+7
      v8h hi = *(const v8h*)(ap + 8);      // channels cb+8..cb+15
      v16h a;
#pragma unroll
      for (int e = 0; e < 8; ++e) { a[e] = lo[e]; a[8 + e] = hi[e]; }
      v16h bb = *(const v16h*)(s_w2 + t * 1024 + oc * 32 + kb);
      acc = __builtin_amdgcn_wmma_f32_16x16x32_f16(false, a, false, bb, (short)0, acc, false, false);
    }
    float bv = cb2[oc];
    int rb = p0 + ((lane & 16) ? 8 : 0);
#pragma unroll
    for (int v = 0; v < 8; ++v)
      s_c2[(rb + v + 1) * 32 + oc] = gelu_f(acc[v] + bv);      // pos p at row p+1
  }
  if (tid < 64) { // zero halo rows p = -1, 512
    int r = (tid >> 5) ? 513 : 0;
    s_c2[r * 32 + (tid & 31)] = 0.0f;
  }
  __syncthreads();

  // stage 3: conv3 + residual + scale
  float scale = scales[ids[b]];
  float c3b = cb3[0];
  for (int p = tid; p < HID; p += 128) {
    float acc = c3b;
#pragma unroll
    for (int c = 0; c < 32; ++c)
#pragma unroll
      for (int t = 0; t < 3; ++t)
        acc += cw3[c * 3 + t] * s_c2[(p + t) * 32 + c];
    out[(size_t)b * HID + p] = (wr[p] + acc) * scale;
  }
}

// ---------------------------------------------------------------------------
extern "C" void kernel_launch(void* const* d_in, const int* in_sizes, int n_in,
                              void* d_out, int out_size, void* d_ws, size_t ws_size,
                              hipStream_t stream) {
  const float* params = (const float*)d_in[0];
  const float* emb    = (const float*)d_in[1];
  const float* W1  = (const float*)d_in[2];
  const float* b1  = (const float*)d_in[3];
  const float* g1  = (const float*)d_in[4];
  const float* be1 = (const float*)d_in[5];
  const float* W2  = (const float*)d_in[6];
  const float* b2  = (const float*)d_in[7];
  const float* g2  = (const float*)d_in[8];
  const float* be2 = (const float*)d_in[9];
  const float* W3  = (const float*)d_in[10];
  const float* b3  = (const float*)d_in[11];
  const float* g3  = (const float*)d_in[12];
  const float* be3 = (const float*)d_in[13];
  const float* Wh  = (const float*)d_in[14];
  const float* bh  = (const float*)d_in[15];
  const float* Wd1 = (const float*)d_in[16];
  const float* bd1 = (const float*)d_in[17];
  const float* Wd2 = (const float*)d_in[18];
  const float* bd2 = (const float*)d_in[19];
  const float* cw1 = (const float*)d_in[20];
  const float* cb1 = (const float*)d_in[21];
  const float* cw2 = (const float*)d_in[22];
  const float* cb2 = (const float*)d_in[23];
  const float* cw3 = (const float*)d_in[24];
  const float* cb3 = (const float*)d_in[25];
  const float* scales = (const float*)d_in[26];
  const int*   ids    = (const int*)d_in[27];
  float* out = (float*)d_out;

  // workspace carving (~83 MB total)
  char* ws = (char*)d_ws;
  size_t off = 0;
  auto carve = [&](size_t bytes) { void* p = ws + off; off += (bytes + 255) & ~(size_t)255; return p; };
  _Float16* x16    = (_Float16*)carve((size_t)NB * 64 * 2);
  _Float16* f16buf = (_Float16*)carve((size_t)NB * 1024 * 2);
  _Float16* hsel16 = (_Float16*)carve((size_t)NB * 512 * 2);
  float*    f32buf = (float*)carve((size_t)NB * 1024 * 4);
  float*    wvec   = (float*)carve((size_t)NB * 512 * 4);
  _Float16* W1t    = (_Float16*)carve((size_t)512 * 64 * 2);
  _Float16* W2t    = (_Float16*)carve((size_t)512 * 512 * 2);
  _Float16* W3t    = (_Float16*)carve((size_t)1024 * 512 * 2);
  _Float16* WhT    = (_Float16*)carve((size_t)3 * 512 * 1024 * 2);
  _Float16* Wd1t   = (_Float16*)carve((size_t)512 * 512 * 2);
  _Float16* Wd2t   = (_Float16*)carve((size_t)512 * 512 * 2);

  auto tlaunch = [&](const float* W, _Float16* Wt, int K, int N, int Kpad) {
    int total = N * Kpad;
    w_to_f16t<<<(total + 255) / 256, 256, 0, stream>>>(W, Wt, K, N, Kpad);
  };
  tlaunch(W1, W1t, 38, 512, 64);
  tlaunch(W2, W2t, 512, 512, 512);
  tlaunch(W3, W3t, 512, 1024, 512);
  for (int e = 0; e < 3; ++e)
    tlaunch(Wh + (size_t)e * 1024 * 512, WhT + (size_t)e * 512 * 1024, 1024, 512, 1024);
  tlaunch(Wd1, Wd1t, 512, 512, 512);
  tlaunch(Wd2, Wd2t, 512, 512, 512);

  prep_kernel<<<NB / 256, 256, 0, stream>>>(params, emb, ids, x16);

  dim3 blk(256);
  dim3 g512(512 / 64, NB / 128);
  dim3 g1024(1024 / 64, NB / 128);

  // layer 1: x @ W1 (+b1) -> LN -> gelu -> f16
  gemm_f16_wmma<<<g512, blk, (size_t)64 * 64 * 2, stream>>>(x16, W1t, b1, f32buf, NB, 512, 64, 64);
  ln_gelu_f16<<<NB, blk, 0, stream>>>(f32buf, g1, be1, f16buf, 512);
  // layer 2
  gemm_f16_wmma<<<g512, blk, (size_t)64 * 512 * 2, stream>>>(f16buf, W2t, b2, f32buf, NB, 512, 512, 512);
  ln_gelu_f16<<<NB, blk, 0, stream>>>(f32buf, g2, be2, f16buf, 512);
  // layer 3 -> feat (B x 1024)
  gemm_f16_wmma<<<g1024, blk, (size_t)64 * 512 * 2, stream>>>(f16buf, W3t, b3, f32buf, NB, 1024, 512, 512);
  ln_gelu_f16<<<NB, blk, 0, stream>>>(f32buf, g3, be3, f16buf, 1024);
  // per-topology head with masked epilogue (B tile = 64x1024 f16 = 128 KB LDS)
  dim3 gh(512 / 64, NB / 128, 3);
  gemm_head_wmma<<<gh, blk, (size_t)64 * 1024 * 2, stream>>>(f16buf, WhT, bh, ids, hsel16);
  // decoder
  gemm_f16_wmma<<<g512, blk, (size_t)64 * 512 * 2, stream>>>(hsel16, Wd1t, bd1, f32buf, NB, 512, 512, 512);
  gelu_to_f16<<<(NB * 512 + 255) / 256, 256, 0, stream>>>(f32buf, f16buf, NB * 512);
  gemm_f16_wmma<<<g512, blk, (size_t)64 * 512 * 2, stream>>>(f16buf, Wd2t, bd2, wvec, NB, 512, 512, 512);
  // fused conv stack + residual + scale
  size_t conv_lds = 65792 + 33024 + 10240 + 1040; // 110096 B < 320 KB/WGP
  conv_fused<<<NB, 128, conv_lds, stream>>>(wvec, cw1, cb1, cw2, cb2, cw3, cb3,
                                            scales, ids, out);
}
